// LSTM_From_Scratch_18210661335500
// MI455X (gfx1250) — compile-verified, matrix operand order checked by reading, not run on hardware
//
#include <hip/hip_runtime.h>

typedef __attribute__((ext_vector_type(16))) __bf16 v16bf;
typedef __attribute__((ext_vector_type(8)))  float  v8f;

#define B_ROWS   4096
#define D_IN     1024
#define HID      2048
#define D_TOT    3072          // INPUT + HIDDEN
#define W_ROWS   8192          // 4 gates x HID, gate-major
#define BM       128           // block tile rows (batch)
#define BN       32            // block tile cols per gate
#define BK       32            // K per WMMA step (bf16)
#define KT       (D_TOT / BK)  // 96 K-steps
#define LDSTR    40            // ushort elems per LDS row (80B, padded from 64B)

union Frag { v16bf v; uint4 u[2]; };

__device__ __forceinline__ uint4 pack8_bf16(float4 a, float4 b) {
  union { unsigned short s[8]; uint4 u; } r;
  r.s[0] = __builtin_bit_cast(unsigned short, (__bf16)a.x);
  r.s[1] = __builtin_bit_cast(unsigned short, (__bf16)a.y);
  r.s[2] = __builtin_bit_cast(unsigned short, (__bf16)a.z);
  r.s[3] = __builtin_bit_cast(unsigned short, (__bf16)a.w);
  r.s[4] = __builtin_bit_cast(unsigned short, (__bf16)b.x);
  r.s[5] = __builtin_bit_cast(unsigned short, (__bf16)b.y);
  r.s[6] = __builtin_bit_cast(unsigned short, (__bf16)b.z);
  r.s[7] = __builtin_bit_cast(unsigned short, (__bf16)b.w);
  return r.u;
}

__device__ __forceinline__ float sigmoid_f(float x) { return 1.0f / (1.0f + __expf(-x)); }
__device__ __forceinline__ float tanh_f(float x)    { return 1.0f - 2.0f / (1.0f + __expf(2.0f * x)); }

// gfx1250 async copy: global -> LDS, 16B per lane, GVS addressing
// (SGPR64 base + per-lane 32-bit byte offset), tracked by ASYNCcnt.
__device__ __forceinline__ void async_b128(unsigned lds_byte_addr, unsigned gbl_byte_off,
                                           const unsigned short* base) {
  asm volatile("global_load_async_to_lds_b128 %0, %1, %2"
               :: "v"(lds_byte_addr), "v"(gbl_byte_off),
                  "s"((unsigned long long)base)
               : "memory");
}
__device__ __forceinline__ void wait_async0() {
  asm volatile("s_wait_asynccnt 0x0" ::: "memory");
}

// ---------- pre-pass: build bf16 operand matrices in workspace ----------
// Abf[row][k] = bf16(concat(x_t, h_prev)[row][k]), row-major [4096 x 3072]
__global__ __launch_bounds__(256) void cvt_combined(const float* __restrict__ x,
                                                    const float* __restrict__ h,
                                                    unsigned short* __restrict__ Abf) {
  const int idx = blockIdx.x * 256 + threadIdx.x;     // one thread per 8 elements
  const int row = idx / (D_TOT / 8);
  const int c8  = (idx % (D_TOT / 8)) * 8;
  const float* src = (c8 < D_IN) ? (x + (size_t)row * D_IN + c8)
                                 : (h + (size_t)row * HID + (c8 - D_IN));
  const float4 f0 = *(const float4*)src;
  const float4 f1 = *(const float4*)(src + 4);
  *(uint4*)&Abf[(size_t)row * D_TOT + c8] = pack8_bf16(f0, f1);
}

// Wbf[g*2048 + n][k] = bf16(W_g[n][k]), gate-major [8192 x 3072]
__global__ __launch_bounds__(256) void cvt_weights(const float* __restrict__ Wi,
                                                   const float* __restrict__ Wf,
                                                   const float* __restrict__ Wc,
                                                   const float* __restrict__ Wo,
                                                   unsigned short* __restrict__ Wbf) {
  const int idx = blockIdx.x * 256 + threadIdx.x;
  const int row = idx / (D_TOT / 8);                  // 0..8191 gate-major
  const int c8  = (idx % (D_TOT / 8)) * 8;
  const int gate = row >> 11;
  const int n    = row & (HID - 1);
  const float* Wg = (gate == 0) ? Wi : (gate == 1) ? Wf : (gate == 2) ? Wc : Wo;
  const float* src = Wg + (size_t)n * D_TOT + c8;
  const float4 f0 = *(const float4*)src;
  const float4 f1 = *(const float4*)(src + 4);
  *(uint4*)&Wbf[(size_t)row * D_TOT + c8] = pack8_bf16(f0, f1);
}

// ---------- main fused GEMM + LSTM epilogue ----------
__global__ __launch_bounds__(256) void lstm_wmma_kernel(
    const unsigned short* __restrict__ Abf, const unsigned short* __restrict__ Wbf,
    const float* __restrict__ c_prev,
    const float* __restrict__ biP, const float* __restrict__ bfP,
    const float* __restrict__ bcP, const float* __restrict__ boP,
    float* __restrict__ h_out, float* __restrict__ c_out)
{
  __shared__ __align__(16) unsigned short ldsA[2][BM * LDSTR];       // 128 x 32 bf16 (padded)
  __shared__ __align__(16) unsigned short ldsW[2][4 * BN * LDSTR];   // 4 gates x 32 x 32 bf16

  const int t    = threadIdx.x;
  const int lane = t & 31;
  const int wave = t >> 5;
  const int mBlock = blockIdx.y * BM;
  const int nBlock = blockIdx.x * BN;

  // ---- async staging: 2 threads per tile row, 32B (2 x b128) each ----
  const int pairRow = t >> 1;                 // 0..127
  const int half16  = t & 1;                  // which 16-element K half
  const int wGate   = pairRow >> 5;
  const int wRowIdx = wGate * HID + nBlock + (pairRow & 31);   // row in Wbf

  const unsigned ldsA_base = (unsigned)(unsigned long long)(void*)&ldsA[0][0];
  const unsigned ldsW_base = (unsigned)(unsigned long long)(void*)&ldsW[0][0];
  const unsigned aDstBase = pairRow * (LDSTR * 2u) + half16 * 32u;
  const unsigned aSrcBase = (unsigned)(((size_t)(mBlock + pairRow) * D_TOT + half16 * 16) * 2);
  const unsigned wSrcBase = (unsigned)(((size_t)wRowIdx * D_TOT + half16 * 16) * 2);

  auto issueAsync = [&](int k0, int buf) {
    const unsigned kB = (unsigned)k0 * 2u;
    const unsigned aDst = ldsA_base + (unsigned)buf * (BM * LDSTR * 2u) + aDstBase;
    const unsigned wDst = ldsW_base + (unsigned)buf * (4 * BN * LDSTR * 2u) + aDstBase;
    async_b128(aDst,       aSrcBase + kB,       Abf);
    async_b128(aDst + 16u, aSrcBase + kB + 16u, Abf);
    async_b128(wDst,       wSrcBase + kB,       Wbf);
    async_b128(wDst + 16u, wSrcBase + kB + 16u, Wbf);
  };

  // ---- WMMA fragment addressing (16-bit A/B layout, wave32) ----
  const int frRow = lane & 15;        // matrix row (A) / col (B) held by lane
  const int half  = lane >> 4;        // lanes 16-31 take K chunks {1,3}
  const int e0    = half * 8;
  const int mSub  = (wave & 3) * 32;  // wave's M offset (2 subtiles of 16)
  const int nSub  = (wave >> 2) * 16; // wave's N offset

  v8f acc[2][4];
  #pragma unroll
  for (int s = 0; s < 2; ++s)
    #pragma unroll
    for (int g = 0; g < 4; ++g)
      acc[s][g] = {};

  issueAsync(0, 0);
  wait_async0();
  __syncthreads();

  int cur = 0, nxt = 1;
  #pragma unroll 1          // single body: keeps accumulators pinned, no mov storms
  for (int kt = 0; kt < KT; ++kt) {
    if (kt + 1 < KT) issueAsync((kt + 1) * BK, nxt);   // overlap with WMMAs below

    const unsigned short* As = &ldsA[cur][0];
    const unsigned short* Ws = &ldsW[cur][0];

    Frag a0, a1;
    {
      const unsigned short* p0 = As + (mSub + frRow) * LDSTR + e0;
      a0.u[0] = *(const uint4*)p0;
      a0.u[1] = *(const uint4*)(p0 + 16);
      const unsigned short* p1 = As + (mSub + 16 + frRow) * LDSTR + e0;
      a1.u[0] = *(const uint4*)p1;
      a1.u[1] = *(const uint4*)(p1 + 16);
    }
    #pragma unroll
    for (int g = 0; g < 4; ++g) {
      Frag b;
      const unsigned short* pb = Ws + (g * BN + nSub + frRow) * LDSTR + e0;
      b.u[0] = *(const uint4*)pb;
      b.u[1] = *(const uint4*)(pb + 16);
      acc[0][g] = __builtin_amdgcn_wmma_f32_16x16x32_bf16(
          false, a0.v, false, b.v, (short)0, acc[0][g], false, false);
      acc[1][g] = __builtin_amdgcn_wmma_f32_16x16x32_bf16(
          false, a1.v, false, b.v, (short)0, acc[1][g], false, false);
    }

    wait_async0();      // next buffer fully in LDS (this wave's part)
    __syncthreads();    // all waves' parts

    const int tmp = cur; cur = nxt; nxt = tmp;
  }

  // ---- fused LSTM epilogue: bias + sigmoid/tanh + c/h update ----
  const int col = nBlock + nSub + frRow;
  const float biV = biP[col], bfV = bfP[col], bcV = bcP[col], boV = boP[col];
  #pragma unroll
  for (int s = 0; s < 2; ++s) {
    const int rowBase = mBlock + mSub + s * 16 + half * 8;   // C-layout: VGPR j -> M=j(+8)
    #pragma unroll
    for (int j = 0; j < 8; ++j) {
      const int row = rowBase + j;
      const size_t idx = (size_t)row * HID + col;
      const float gi = acc[s][0][j] + biV;
      const float gf = acc[s][1][j] + bfV;
      const float gc = acc[s][2][j] + bcV;
      const float go = acc[s][3][j] + boV;
      const float it = sigmoid_f(gi);
      const float ft = sigmoid_f(gf);
      const float gt = tanh_f(gc);
      const float ot = sigmoid_f(go);
      const float ct = ft * c_prev[idx] + it * gt;
      h_out[idx] = ot * tanh_f(ct);
      c_out[idx] = ct;
    }
  }
}

extern "C" void kernel_launch(void* const* d_in, const int* in_sizes, int n_in,
                              void* d_out, int out_size, void* d_ws, size_t ws_size,
                              hipStream_t stream) {
  (void)in_sizes; (void)n_in; (void)out_size; (void)ws_size;
  const float* x_t    = (const float*)d_in[0];
  const float* h_prev = (const float*)d_in[1];
  const float* c_prev = (const float*)d_in[2];
  const float* Wi = (const float*)d_in[3];
  const float* bi = (const float*)d_in[4];
  const float* Wf = (const float*)d_in[5];
  const float* bf = (const float*)d_in[6];
  const float* Wc = (const float*)d_in[7];
  const float* bc = (const float*)d_in[8];
  const float* Wo = (const float*)d_in[9];
  const float* bo = (const float*)d_in[10];
  float* h_out = (float*)d_out;
  float* c_out = (float*)d_out + (size_t)B_ROWS * HID;

  unsigned short* Abf = (unsigned short*)d_ws;                     // 25.2 MB
  unsigned short* Wbf = Abf + (size_t)B_ROWS * D_TOT;              // 50.3 MB

  cvt_combined<<<dim3(B_ROWS * (D_TOT / 8) / 256), dim3(256), 0, stream>>>(x_t, h_prev, Abf);
  cvt_weights <<<dim3(W_ROWS * (D_TOT / 8) / 256), dim3(256), 0, stream>>>(Wi, Wf, Wc, Wo, Wbf);

  dim3 grid(HID / BN, B_ROWS / BM);   // (64, 32) = 2048 workgroups
  lstm_wmma_kernel<<<grid, dim3(256), 0, stream>>>(
      Abf, Wbf, c_prev, bi, bf, bc, bo, h_out, c_out);
}
